// ObjectAttentionBlock_34875134443879
// MI455X (gfx1250) — matrix operand order repaired and checked by reading, not compiled, not run on hardware
//
#include <hip/hip_runtime.h>

// ---------------------------------------------------------------------------
// ObjectAttentionBlock fused pipeline for gfx1250 (MI455X), bf16 WMMA path.
// ---------------------------------------------------------------------------

typedef __attribute__((ext_vector_type(16))) __bf16 bf16x16;
typedef __attribute__((ext_vector_type(8)))  __bf16 bf16x8;
typedef __attribute__((ext_vector_type(4)))  __bf16 bf16x4;
typedef __attribute__((ext_vector_type(8)))  float  f32x8;
typedef __attribute__((ext_vector_type(4)))  float  f32x4;

#define WMMA_BF16(C, A, B) \
  (C) = __builtin_amdgcn_wmma_f32_16x16x32_bf16(false, (A), false, (B), (short)0, (C), false, false)

#define HW 16384      // 128*128
#define NB 8          // batch
#define TILE_N 64     // spatial positions per workgroup
#define LDX 520       // LDS row stride (bf16) for 512-wide buffer (+8 pad, 16B aligned rows)
#define LDQ 264       // LDS row stride (bf16) for 256-wide buffer

// LDS byte offsets
#define SM_A    0                       // 64 x LDX bf16 = 66560 B (x tile, then q tile)
#define SM_B    66560                   // 64 x LDQ bf16 = 33792 B (q1 tile, then context)
#define SM_SF   100352                  // 64 x 32 f32   = 8192  B (sim logits)
#define SM_PB   108544                  // 64 x 32 bf16  = 4096  B (softmax probs)
#define SM_BIAS 112640                  // 1024 f32      = 4096  B (b1|b2|b3)
#define SMEM_BYTES 116736

// Workspace layout (bytes)
#define OFF_W1   0u                               // 256x512 bf16
#define OFF_W2   (OFF_W1  + 256u*512u*2u)         // 256x256 bf16
#define OFF_W3   (OFF_W2  + 256u*256u*2u)         // 512x256 bf16
#define OFF_B1   (OFF_W3  + 512u*256u*2u)
#define OFF_B2   (OFF_B1  + 256u*4u)
#define OFF_B3   (OFF_B2  + 256u*4u)
#define OFF_WO1  (OFF_B3  + 512u*4u)              // 256x512 f32
#define OFF_WO2  (OFF_WO1 + 256u*512u*4u)         // 256x256 f32
#define OFF_WD   (OFF_WO2 + 256u*256u*4u)         // 256x512 f32
#define OFF_BO1  (OFF_WD  + 256u*512u*4u)
#define OFF_BO2  (OFF_BO1 + 256u*4u)
#define OFF_BD   (OFF_BO2 + 256u*4u)
#define OFF_KEY  (OFF_BD  + 256u*4u)              // 8 x 32 x 256 bf16 (rows >=19 zero)
#define OFF_VALT (OFF_KEY + 8u*32u*256u*2u)       // 8 x 256 x 32 bf16 (cols >=19 zero)

// ---------------------------------------------------------------------------
// Fragment loaders (ISA 7.12.2 layouts, wave32)
// A 16x32 bf16: lane L -> row M = L&15; K chunks [Kb,Kb+8) and [Kb+16,Kb+24), Kb=(L>>4)*8
// B 32x16 bf16: lane L -> col N = L&15; K chunks [Kb,Kb+8) and [Kb+16,Kb+24), Kb=(L>>4)*8
// C/D 16x16 f32: lane L -> col N = L&15; rows M = v + (L>>4)*8, v=0..7
// ---------------------------------------------------------------------------
__device__ __forceinline__ bf16x16 load_a_frag(const __bf16* __restrict__ W, int row_base,
                                               int ldk, int k0, int lane) {
  const int r  = row_base + (lane & 15);
  const int kb = k0 + ((lane >> 4) << 3);
  const __bf16* p = W + (size_t)r * ldk + kb;
  union { bf16x16 v; bf16x8 h[2]; } u;
  u.h[0] = *(const bf16x8*)(p);
  u.h[1] = *(const bf16x8*)(p + 16);
  return u.v;
}

__device__ __forceinline__ bf16x16 load_b_frag(const __bf16* L, int n_base, int ld,
                                               int k0, int lane) {
  const int n  = n_base + (lane & 15);
  const int ks = k0 + ((lane >> 4) << 3);
  const __bf16* p = L + n * ld + ks;
  union { bf16x16 v; bf16x8 h[2]; } u;
  u.h[0] = *(const bf16x8*)(p);
  u.h[1] = *(const bf16x8*)(p + 16);
  return u.v;
}

// Compile-time bias/relu selection: no runtime-null branches in the epilogue.
template <bool HAS_BIAS, bool RELU>
__device__ __forceinline__ void store_tile_lds(__bf16* dst, int ld, int n_base, int c_base,
                                               const float* bias, f32x8 acc, int lane) {
  const int n  = n_base + (lane & 15);
  const int c0 = c_base + ((lane >> 4) << 3);
  float bb[8];
  if (HAS_BIAS) {
    const f32x4 b0 = *(const f32x4*)(bias + c0);      // c0 % 8 == 0 -> 16B aligned
    const f32x4 b1 = *(const f32x4*)(bias + c0 + 4);
#pragma unroll
    for (int i = 0; i < 4; ++i) { bb[i] = b0[i]; bb[4 + i] = b1[i]; }
  }
  union { bf16x8 v; __bf16 e[8]; } u;
#pragma unroll
  for (int i = 0; i < 8; ++i) {
    float x = acc[i];
    if (HAS_BIAS) x += bb[i];
    if (RELU) x = fmaxf(x, 0.0f);
    u.e[i] = (__bf16)x;
  }
  *(bf16x8*)(dst + n * ld + c0) = u.v;
}

// ---------------------------------------------------------------------------
// Prep kernel 1: fold BN into conv weights/bias. W_eff = W * s * oscale,
// b_eff = oscale*(b*s + be - m*s), s = g/sqrt(v+eps). bf16 and/or f32 output.
// ---------------------------------------------------------------------------
__global__ void fold_kernel(const float* __restrict__ w, const float* __restrict__ b,
                            const float* __restrict__ g, const float* __restrict__ be,
                            const float* __restrict__ m, const float* __restrict__ v,
                            int cin, __bf16* wbf, float* wf, float* bout, float oscale) {
  const int o = blockIdx.x;
  const float s0 = g[o] * rsqrtf(v[o] + 1e-5f);
  const float s  = s0 * oscale;
  for (int i = threadIdx.x; i < cin; i += blockDim.x) {
    const float we = w[(size_t)o * cin + i] * s;
    if (wbf) wbf[(size_t)o * cin + i] = (__bf16)we;
    if (wf)  wf[(size_t)o * cin + i]  = we;
  }
  if (threadIdx.x == 0)
    bout[o] = oscale * (b[o] * s0 + be[o] - m[o] * s0);
}

// ---------------------------------------------------------------------------
// Prep kernel 2: key/value from proxy (tiny). One block per batch, 256 thr.
//   key_pad[b][kp][c]  (kp padded to 32, softmax scale folded into fo2)
//   valueT_pad[b][c][kp] (kp padded to 32)
// ---------------------------------------------------------------------------
__global__ void kv_kernel(const float* __restrict__ proxy,
                          const float* __restrict__ Wo1, const float* __restrict__ bo1,
                          const float* __restrict__ Wo2, const float* __restrict__ bo2,
                          const float* __restrict__ Wd,  const float* __restrict__ bd,
                          __bf16* __restrict__ keypad, __bf16* __restrict__ valT) {
  __shared__ float t1[19][256];
  const int b  = blockIdx.x;
  const int co = threadIdx.x;            // 0..255
  const float* px = proxy + (size_t)b * 512 * 19;

  for (int kp = 0; kp < 19; ++kp) {
    float acc = bo1[co];
    for (int i = 0; i < 512; ++i) acc += Wo1[(size_t)co * 512 + i] * px[i * 19 + kp];
    t1[kp][co] = fmaxf(acc, 0.0f);
  }
  __syncthreads();

  for (int kp = 0; kp < 32; ++kp) {
    float val = 0.0f;
    if (kp < 19) {
      float acc = bo2[co];
      for (int i = 0; i < 256; ++i) acc += Wo2[(size_t)co * 256 + i] * t1[kp][i];
      val = fmaxf(acc, 0.0f);            // 1/sqrt(kc) already folded into Wo2/bo2
    }
    keypad[((size_t)b * 32 + kp) * 256 + co] = (__bf16)val;
  }

  float vv[19];
  for (int kp = 0; kp < 19; ++kp) {
    float acc = bd[co];
    for (int i = 0; i < 512; ++i) acc += Wd[(size_t)co * 512 + i] * px[i * 19 + kp];
    vv[kp] = fmaxf(acc, 0.0f);
  }
  for (int kp = 0; kp < 32; ++kp)
    valT[((size_t)b * 256 + co) * 32 + kp] = (__bf16)(kp < 19 ? vv[kp] : 0.0f);
}

// ---------------------------------------------------------------------------
// Main fused kernel: per workgroup = one batch b, 64 spatial positions.
// 8 waves / 256 threads. All matmuls via v_wmma_f32_16x16x32_bf16.
// ---------------------------------------------------------------------------
__global__ __launch_bounds__(256, 1) void main_kernel(
    const float* __restrict__ x,
    const __bf16* __restrict__ W1, const __bf16* __restrict__ W2,
    const __bf16* __restrict__ W3,
    const float* __restrict__ b1, const float* __restrict__ b2,
    const float* __restrict__ b3,
    const __bf16* __restrict__ keypad, const __bf16* __restrict__ valT,
    float* __restrict__ out) {
  extern __shared__ char smem[];
  __bf16* A    = (__bf16*)(smem + SM_A);     // x tile (64 x LDX), later q tile (64 x LDQ)
  __bf16* Bq   = (__bf16*)(smem + SM_B);     // q1 tile, later context tile (64 x LDQ)
  float*  Sf   = (float*) (smem + SM_SF);    // sim logits 64 x 32
  __bf16* Pb   = (__bf16*)(smem + SM_PB);    // softmax probs 64 x 32
  float*  bias = (float*) (smem + SM_BIAS);  // b1[0..256) b2[256..512) b3[512..1024)

  const int tid  = threadIdx.x;
  const int lane = tid & 31;
  const int wave = tid >> 5;
  const int blk  = blockIdx.x;
  const int b    = blk >> 8;                 // 256 tiles per batch image
  const int hw0  = (blk & 255) * TILE_N;

  // ---- stage biases -------------------------------------------------------
  bias[tid]       = b1[tid];
  bias[256 + tid] = b2[tid];
  bias[512 + tid]       = b3[tid];
  bias[512 + 256 + tid] = b3[256 + tid];

  // ---- stage x tile: [c][n] f32 (HBM) -> [n][c] bf16 (LDS) ---------------
  {
    const int n  = tid & 63;
    const int cg = tid >> 6;                 // 0..3
    const float* xb = x + (size_t)b * 512 * HW + hw0 + n;
    __bf16* dst = A + n * LDX;
#pragma unroll 4
    for (int it = 0; it < 32; ++it) {
      const int c = it * 16 + cg * 4;
      const float f0 = __builtin_nontemporal_load(xb + (size_t)(c + 0) * HW);
      const float f1 = __builtin_nontemporal_load(xb + (size_t)(c + 1) * HW);
      const float f2 = __builtin_nontemporal_load(xb + (size_t)(c + 2) * HW);
      const float f3 = __builtin_nontemporal_load(xb + (size_t)(c + 3) * HW);
      union { bf16x4 v; __bf16 e[4]; } u;
      u.e[0] = (__bf16)f0; u.e[1] = (__bf16)f1;
      u.e[2] = (__bf16)f2; u.e[3] = (__bf16)f3;
      *(bf16x4*)(dst + c) = u.v;
    }
  }
  __syncthreads();

  // ---- GEMM1: q1 = relu(W1 @ x + b1)   [256 x 64], K=512 ------------------
  {
    f32x8 acc[2][4] = {};
    for (int kk = 0; kk < 16; ++kk) {
      const int k0 = kk * 32;
      bf16x16 bf[4];
#pragma unroll
      for (int nt = 0; nt < 4; ++nt) bf[nt] = load_b_frag(A, nt * 16, LDX, k0, lane);
#pragma unroll
      for (int mt = 0; mt < 2; ++mt) {
        const bf16x16 af = load_a_frag(W1, (wave * 2 + mt) * 16, 512, k0, lane);
#pragma unroll
        for (int nt = 0; nt < 4; ++nt) WMMA_BF16(acc[mt][nt], af, bf[nt]);
      }
    }
#pragma unroll
    for (int mt = 0; mt < 2; ++mt)
#pragma unroll
      for (int nt = 0; nt < 4; ++nt)
        store_tile_lds<true, true>(Bq, LDQ, nt * 16, (wave * 2 + mt) * 16, bias,
                                   acc[mt][nt], lane);
  }
  __syncthreads();

  // ---- GEMM2: q = relu(W2 @ q1 + b2)   [256 x 64], K=256 -> region A ------
  {
    f32x8 acc[2][4] = {};
    for (int kk = 0; kk < 8; ++kk) {
      const int k0 = kk * 32;
      bf16x16 bf[4];
#pragma unroll
      for (int nt = 0; nt < 4; ++nt) bf[nt] = load_b_frag(Bq, nt * 16, LDQ, k0, lane);
#pragma unroll
      for (int mt = 0; mt < 2; ++mt) {
        const bf16x16 af = load_a_frag(W2, (wave * 2 + mt) * 16, 256, k0, lane);
#pragma unroll
        for (int nt = 0; nt < 4; ++nt) WMMA_BF16(acc[mt][nt], af, bf[nt]);
      }
    }
    __syncthreads();   // all waves done reading A (x tile) before overwrite
#pragma unroll
    for (int mt = 0; mt < 2; ++mt)
#pragma unroll
      for (int nt = 0; nt < 4; ++nt)
        store_tile_lds<true, true>(A, LDQ, nt * 16, (wave * 2 + mt) * 16, bias + 256,
                                   acc[mt][nt], lane);
  }
  __syncthreads();

  // ---- sim = key_pad @ q   [32(19) x 64], K=256 ---------------------------
  {
    const int mt = wave & 1;                 // 2 M-tiles
    const int nt = wave >> 1;                // 4 N-tiles
    f32x8 acc = {};
    const __bf16* kb = keypad + (size_t)b * 32 * 256;
    for (int kk = 0; kk < 8; ++kk) {
      const int k0 = kk * 32;
      const bf16x16 bf = load_b_frag(A, nt * 16, LDQ, k0, lane);
      const bf16x16 af = load_a_frag(kb, mt * 16, 256, k0, lane);
      WMMA_BF16(acc, af, bf);
    }
    const int n  = nt * 16 + (lane & 15);
    const int kr = mt * 16 + ((lane >> 4) << 3);
#pragma unroll
    for (int i = 0; i < 8; ++i) Sf[n * 32 + kr + i] = acc[i];
  }
  __syncthreads();

  // ---- softmax over 19 keys (rows >=19 of Sf are exact zeros, masked) -----
  if (tid < 64) {
    float srow[19];
    float mx = -1e30f;
#pragma unroll
    for (int k = 0; k < 19; ++k) { srow[k] = Sf[tid * 32 + k]; mx = fmaxf(mx, srow[k]); }
    float ssum = 0.0f;
#pragma unroll
    for (int k = 0; k < 19; ++k) { const float e = __expf(srow[k] - mx); srow[k] = e; ssum += e; }
    const float inv = __frcp_rn(ssum);
#pragma unroll
    for (int k = 0; k < 32; ++k)
      Pb[tid * 32 + k] = (__bf16)(k < 19 ? srow[k] * inv : 0.0f);
  }
  __syncthreads();

  // ---- context = valueT @ P   [256 x 64], K=32 (single step) -> region Bq -
  {
    f32x8 acc[2][4] = {};
    bf16x16 bf[4];
#pragma unroll
    for (int nt = 0; nt < 4; ++nt) bf[nt] = load_b_frag(Pb, nt * 16, 32, 0, lane);
    const __bf16* vb = valT + (size_t)b * 256 * 32;
#pragma unroll
    for (int mt = 0; mt < 2; ++mt) {
      const bf16x16 af = load_a_frag(vb, (wave * 2 + mt) * 16, 32, 0, lane);
#pragma unroll
      for (int nt = 0; nt < 4; ++nt) WMMA_BF16(acc[mt][nt], af, bf[nt]);
    }
#pragma unroll
    for (int mt = 0; mt < 2; ++mt)
#pragma unroll
      for (int nt = 0; nt < 4; ++nt)
        store_tile_lds<false, false>(Bq, LDQ, nt * 16, (wave * 2 + mt) * 16, nullptr,
                                     acc[mt][nt], lane);
  }
  __syncthreads();

  // ---- GEMM3: out = relu(W3 @ context + b3)   [512 x 64], K=256 -----------
  {
    f32x8 acc[4][4] = {};
    for (int kk = 0; kk < 8; ++kk) {
      const int k0 = kk * 32;
      bf16x16 bf[4];
#pragma unroll
      for (int nt = 0; nt < 4; ++nt) bf[nt] = load_b_frag(Bq, nt * 16, LDQ, k0, lane);
#pragma unroll
      for (int mt = 0; mt < 4; ++mt) {
        const bf16x16 af = load_a_frag(W3, (wave * 4 + mt) * 16, 256, k0, lane);
#pragma unroll
        for (int nt = 0; nt < 4; ++nt) WMMA_BF16(acc[mt][nt], af, bf[nt]);
      }
    }
    // epilogue: bias + relu, NT stores straight to HBM in [b][c][hw] order
#pragma unroll
    for (int mt = 0; mt < 4; ++mt) {
      const int c0 = (wave * 4 + mt) * 16 + ((lane >> 4) << 3);
      const f32x4 bb0 = *(const f32x4*)(bias + 512 + c0);      // 16B-aligned LDS loads
      const f32x4 bb1 = *(const f32x4*)(bias + 512 + c0 + 4);
      float bb[8];
#pragma unroll
      for (int i = 0; i < 4; ++i) { bb[i] = bb0[i]; bb[4 + i] = bb1[i]; }
#pragma unroll
      for (int nt = 0; nt < 4; ++nt) {
        const int n = hw0 + nt * 16 + (lane & 15);
        float* ob = out + (size_t)b * 512 * HW + n;
#pragma unroll
        for (int i = 0; i < 8; ++i) {
          const float v = fmaxf(acc[mt][nt][i] + bb[i], 0.0f);
          __builtin_nontemporal_store(v, ob + (size_t)(c0 + i) * HW);
        }
      }
    }
  }
}

// ---------------------------------------------------------------------------
extern "C" void kernel_launch(void* const* d_in, const int* in_sizes, int n_in,
                              void* d_out, int out_size, void* d_ws, size_t ws_size,
                              hipStream_t stream) {
  (void)in_sizes; (void)n_in; (void)out_size; (void)ws_size;
  const float* x     = (const float*)d_in[0];
  const float* proxy = (const float*)d_in[1];
  // per-layer params: w,b,g,be,m,v starting at index 2, layer order fp1,fp2,fo1,fo2,fd,fu
  const float* P[6][6];
  for (int l = 0; l < 6; ++l)
    for (int j = 0; j < 6; ++j)
      P[l][j] = (const float*)d_in[2 + l * 6 + j];

  char* ws = (char*)d_ws;
  __bf16* W1   = (__bf16*)(ws + OFF_W1);
  __bf16* W2   = (__bf16*)(ws + OFF_W2);
  __bf16* W3   = (__bf16*)(ws + OFF_W3);
  float*  b1   = (float*) (ws + OFF_B1);
  float*  b2   = (float*) (ws + OFF_B2);
  float*  b3   = (float*) (ws + OFF_B3);
  float*  Wo1  = (float*) (ws + OFF_WO1);
  float*  Wo2  = (float*) (ws + OFF_WO2);
  float*  Wd   = (float*) (ws + OFF_WD);
  float*  bo1  = (float*) (ws + OFF_BO1);
  float*  bo2  = (float*) (ws + OFF_BO2);
  float*  bd   = (float*) (ws + OFF_BD);
  __bf16* keyp = (__bf16*)(ws + OFF_KEY);
  __bf16* valT = (__bf16*)(ws + OFF_VALT);

  const float kscale = 0.0625f;  // 1/sqrt(256), folded into fo2

  // fp1: 512->256 (bf16)
  fold_kernel<<<256, 256, 0, stream>>>(P[0][0], P[0][1], P[0][2], P[0][3], P[0][4], P[0][5],
                                       512, W1, nullptr, b1, 1.0f);
  // fp2: 256->256 (bf16)
  fold_kernel<<<256, 256, 0, stream>>>(P[1][0], P[1][1], P[1][2], P[1][3], P[1][4], P[1][5],
                                       256, W2, nullptr, b2, 1.0f);
  // fo1: 512->256 (f32, proxy path)
  fold_kernel<<<256, 256, 0, stream>>>(P[2][0], P[2][1], P[2][2], P[2][3], P[2][4], P[2][5],
                                       512, nullptr, Wo1, bo1, 1.0f);
  // fo2: 256->256 (f32, softmax scale folded in)
  fold_kernel<<<256, 256, 0, stream>>>(P[3][0], P[3][1], P[3][2], P[3][3], P[3][4], P[3][5],
                                       256, nullptr, Wo2, bo2, kscale);
  // fd: 512->256 (f32)
  fold_kernel<<<256, 256, 0, stream>>>(P[4][0], P[4][1], P[4][2], P[4][3], P[4][4], P[4][5],
                                       512, nullptr, Wd, bd, 1.0f);
  // fu: 256->512 (bf16)
  fold_kernel<<<512, 256, 0, stream>>>(P[5][0], P[5][1], P[5][2], P[5][3], P[5][4], P[5][5],
                                       256, W3, nullptr, b3, 1.0f);

  kv_kernel<<<NB, 256, 0, stream>>>(proxy, Wo1, bo1, Wo2, bo2, Wd, bd, keyp, valT);

  main_kernel<<<NB * (HW / TILE_N), 256, SMEM_BYTES, stream>>>(
      x, W1, W2, W3, b1, b2, b3, keyp, valT, (float*)d_out);
}
